// FusedTriangleMultiplicationOpti_11991548690620
// MI455X (gfx1250) — matrix operand order repaired and checked by reading, not compiled, not run on hardware
//
#include <hip/hip_runtime.h>
#include <hip/hip_bf16.h>

// ---------------------------------------------------------------------------
// AF2 TriangleMultiplication (outgoing) for MI455X (gfx1250, wave32, WMMA).
// All GEMM-shaped work runs on v_wmma_f32_16x16x32_bf16 (bf16 operands, f32
// accum).  The einsum kernel (dominant, 116 GFLOP) streams its K-slabs with
// GLOBAL_LOAD_ASYNC_TO_LDS_B128 + ASYNCcnt double buffering so the Tensor/WMMA
// pipe overlaps the DMA.
// Workspace layout (bytes), S = 768*768*128*2 = 150,994,944:
//   [0,S)    x_bf16[p][128]       LN(act), bf16
//   [S,2S)   left_t[c][i][k]      bf16
//   [2S,3S)  right_t[c][j][k]     bf16
//   [3S,4S)  gate_t[o][p]         sigmoid(gating linear), bf16
//   [4S,6S)  out_t[c][i][j]       f32 einsum result
//   [6S,..)  packed weights (Wcat bf16 640x128, bias 640 f32, Wo bf16 128x128)
// Total ~906 MB.
// ---------------------------------------------------------------------------

typedef __attribute__((ext_vector_type(16))) __bf16 bf16x16;
typedef __attribute__((ext_vector_type(8)))  __bf16 bf16x8;
typedef __attribute__((ext_vector_type(4)))  __bf16 bf16x4;
typedef __attribute__((ext_vector_type(8)))  float  f32x8;

#define NRES 768
#define CZ   128
#define NPOS (768 * 768)

static __device__ __forceinline__ f32x8 zero8() {
  f32x8 z;
#pragma unroll
  for (int i = 0; i < 8; ++i) z[i] = 0.0f;
  return z;
}

static __device__ __forceinline__ bf16x16 frag16(const __bf16* p0, const __bf16* p1) {
  bf16x8 lo = *(const bf16x8*)p0;
  bf16x8 hi = *(const bf16x8*)p1;
  return __builtin_shufflevector(lo, hi, 0, 1, 2, 3, 4, 5, 6, 7,
                                 8, 9, 10, 11, 12, 13, 14, 15);
}

// A fragment (16x32 bf16): lane holds row (l&15); K chunks [kb,kb+8) and
// [kb+16,kb+24) with kb = (l>>4)*8.  row_ptr points at row start (k=ks*32).
static __device__ __forceinline__ bf16x16 load_fragA(const __bf16* row_ptr, int lane) {
  int kb = (lane >> 4) * 8;
  return frag16(row_ptr + kb, row_ptr + kb + 16);
}

// B fragment (32x16, K x N): lane holds column (l&15) = row of B^T; K chunks
// [kb,kb+8) and [kb+16,kb+24) with kb = (l>>4)*8.
static __device__ __forceinline__ bf16x16 load_fragB(const __bf16* row_ptr, int lane) {
  int kb = (lane >> 4) * 8;
  return frag16(row_ptr + kb, row_ptr + kb + 16);
}

static __device__ __forceinline__ f32x8 wmma_bf16(bf16x16 a, bf16x16 b, f32x8 c) {
  return __builtin_amdgcn_wmma_f32_16x16x32_bf16(false, a, false, b,
                                                 (short)0, c, false, false);
}

static __device__ __forceinline__ float sigf(float x) {
  return 1.0f / (1.0f + __expf(-x));
}

static __device__ __forceinline__ float wred32(float v) {
#pragma unroll
  for (int m = 16; m >= 1; m >>= 1) v += __shfl_xor(v, m, 32);
  return v;
}

// Async DMA: 16 bytes global -> LDS, tracked by ASYNCcnt.  lds_off is the
// wave-relative LDS byte address (low 32 bits of the generic pointer, per
// ISA flat->LDS truncation).
static __device__ __forceinline__ void async_ld16(const __bf16* lds_ptr,
                                                  const __bf16* gptr) {
  unsigned lds_off = (unsigned)(unsigned long long)(uintptr_t)lds_ptr;
  asm volatile("global_load_async_to_lds_b128 %0, %1, off"
               :
               : "v"(lds_off), "v"((unsigned long long)(uintptr_t)gptr)
               : "memory");
}

static __device__ __forceinline__ void wait_async0() {
  asm volatile("s_wait_asynccnt 0x0" ::: "memory");
}

// ---------------------------------------------------------------------------
// Kernel 0: pack weights -> bf16.  Wcat rows interleaved in quadruples:
//   m = 4c+0: proj row c       (left proj,  channel c)
//   m = 4c+1: proj row 128+c   (right proj, channel c)
//   m = 4c+2: gate row c       (left gate)
//   m = 4c+3: gate row 128+c   (right gate)
//   m = 512+o: gating_linear row o
// ---------------------------------------------------------------------------
__global__ void k0_pack(const float* pw, const float* pb,
                        const float* gw, const float* gb,
                        const float* glw, const float* glb,
                        const float* ow,
                        __bf16* wcat, float* bpk, __bf16* wo) {
  int idx = blockIdx.x * 256 + threadIdx.x;
  if (idx < 640 * 128) {
    int m = idx >> 7, k = idx & 127;
    float v;
    if (m < 512) {
      int c = m >> 2, t = m & 3;
      v = (t == 0) ? pw[c * 128 + k]
        : (t == 1) ? pw[(128 + c) * 128 + k]
        : (t == 2) ? gw[c * 128 + k]
                   : gw[(128 + c) * 128 + k];
    } else {
      v = glw[(m - 512) * 128 + k];
    }
    wcat[m * 128 + k] = (__bf16)v;
    if (k == 0) {
      float b;
      if (m < 512) {
        int c = m >> 2, t = m & 3;
        b = (t == 0) ? pb[c] : (t == 1) ? pb[128 + c]
          : (t == 2) ? gb[c] : gb[128 + c];
      } else {
        b = glb[m - 512];
      }
      bpk[m] = b;
    }
  } else if (idx < 640 * 128 + 128 * 128) {
    int j = idx - 640 * 128;
    wo[j] = (__bf16)ow[j];
  }
}

// ---------------------------------------------------------------------------
// Kernel 1: input LayerNorm, one wave per position p.  x_bf16[p][128].
// ---------------------------------------------------------------------------
__global__ void k1_layernorm(const float* __restrict__ act,
                             const float* __restrict__ lnw,
                             const float* __restrict__ lnb,
                             __bf16* __restrict__ x) {
  int lane = threadIdx.x & 31;
  size_t p = (size_t)blockIdx.x * 8 + (threadIdx.x >> 5);
  const float4 v = *(const float4*)(act + p * 128 + lane * 4);
  float s = wred32(v.x + v.y + v.z + v.w);
  float mu = s * (1.0f / 128.0f);
  float d0 = v.x - mu, d1 = v.y - mu, d2 = v.z - mu, d3 = v.w - mu;
  float ss = wred32(d0 * d0 + d1 * d1 + d2 * d2 + d3 * d3);
  float rstd = rsqrtf(ss * (1.0f / 128.0f) + 1e-5f);
  const float4 w4 = *(const float4*)(lnw + lane * 4);
  const float4 b4 = *(const float4*)(lnb + lane * 4);
  bf16x4 o;
  o[0] = (__bf16)(d0 * rstd * w4.x + b4.x);
  o[1] = (__bf16)(d1 * rstd * w4.y + b4.y);
  o[2] = (__bf16)(d2 * rstd * w4.z + b4.z);
  o[3] = (__bf16)(d3 * rstd * w4.w + b4.w);
  *(bf16x4*)(x + p * 128 + lane * 4) = o;
}

// ---------------------------------------------------------------------------
// Kernel 2: C[m][p] = Wcat[m][:] . x[p][:], M=640, K=128, N-tile = 128 pos.
// Fused epilogue: left/right = mask*(proj+b)*sigmoid(gate+b) -> bf16,
// gate_t = sigmoid(gating+b) -> bf16.  8 waves, each owns 16 positions.
// ---------------------------------------------------------------------------
__global__ void k2_project(const __bf16* __restrict__ x,
                           const __bf16* __restrict__ wcat,
                           const float* __restrict__ bpk,
                           const float* __restrict__ mask,
                           __bf16* __restrict__ left,
                           __bf16* __restrict__ right,
                           __bf16* __restrict__ gate) {
  extern __shared__ __align__(16) char smem[];
  __bf16* sA = (__bf16*)smem;          // 160 x 136 (chunk of Wcat)
  __bf16* sX = sA + 160 * 136;         // 128 x 136 (x tile)

  const int t = threadIdx.x, lane = t & 31, w = t >> 5;
  const size_t p0 = (size_t)blockIdx.x * 128;

  // stage x tile: 128 rows x 16 chunks of 8 bf16
#pragma unroll
  for (int it = 0; it < 8; ++it) {
    int cidx = it * 256 + t;
    int row = cidx >> 4, q = cidx & 15;
    *(bf16x8*)(sX + row * 136 + q * 8) =
        *(const bf16x8*)(x + (p0 + row) * 128 + q * 8);
  }
  __syncthreads();

  // persistent B fragments: this wave's 16 positions, 4 K-steps
  bf16x16 bfr[4];
#pragma unroll
  for (int ks = 0; ks < 4; ++ks)
    bfr[ks] = load_fragB(sX + (w * 16 + (lane & 15)) * 136 + ks * 32, lane);

  const int pmy = (int)p0 + w * 16 + (lane & 15);
  const float maskv = mask[pmy];
  const int h8 = (lane >> 4) * 8;

  for (int ch = 0; ch < 4; ++ch) {
    __syncthreads();
    // stage 160 rows of Wcat: 160 x 16 chunks
#pragma unroll
    for (int it = 0; it < 10; ++it) {
      int cidx = it * 256 + t;
      int row = cidx >> 4, q = cidx & 15;
      *(bf16x8*)(sA + row * 136 + q * 8) =
          *(const bf16x8*)(wcat + (ch * 160 + row) * 128 + q * 8);
    }
    __syncthreads();

#pragma unroll
    for (int msl = 0; msl < 10; ++msl) {
      const int m0 = ch * 160 + msl * 16;
      f32x8 acc = zero8();
#pragma unroll
      for (int ks = 0; ks < 4; ++ks) {
        bf16x16 a = load_fragA(sA + (msl * 16 + (lane & 15)) * 136 + ks * 32, lane);
        acc = wmma_bf16(a, bfr[ks], acc);
      }
      if (m0 < 512) {
        const int base = m0 + h8;
        const int c0 = base >> 2;
        float l0 = (acc[0] + bpk[base + 0]) * sigf(acc[2] + bpk[base + 2]) * maskv;
        float r0 = (acc[1] + bpk[base + 1]) * sigf(acc[3] + bpk[base + 3]) * maskv;
        float l1 = (acc[4] + bpk[base + 4]) * sigf(acc[6] + bpk[base + 6]) * maskv;
        float r1 = (acc[5] + bpk[base + 5]) * sigf(acc[7] + bpk[base + 7]) * maskv;
        left[(size_t)c0 * NPOS + pmy] = (__bf16)l0;
        left[(size_t)(c0 + 1) * NPOS + pmy] = (__bf16)l1;
        right[(size_t)c0 * NPOS + pmy] = (__bf16)r0;
        right[(size_t)(c0 + 1) * NPOS + pmy] = (__bf16)r1;
      } else {
#pragma unroll
        for (int v = 0; v < 8; ++v) {
          int m = m0 + h8 + v;
          gate[(size_t)(m - 512) * NPOS + pmy] = (__bf16)sigf(acc[v] + bpk[m]);
        }
      }
    }
  }
}

// ---------------------------------------------------------------------------
// Kernel 3: einsum 'ikc,jkc->ijc'.  Per block: channel c, 128x128 (i,j) tile,
// K = 768 in steps of 32.  8 waves = 2(i) x 4(j); each wave 64x32 = 4x2
// subtiles.  K-slabs stream via GLOBAL_LOAD_ASYNC_TO_LDS_B128 into
// double-buffered LDS tiles (row stride 40 bf16 for bank decorrelation);
// s_wait_asynccnt + barrier closes each stage while the next slab is in
// flight behind the 8-WMMA compute body.
// ---------------------------------------------------------------------------
__global__ void k3_einsum(const __bf16* __restrict__ left,
                          const __bf16* __restrict__ right,
                          float* __restrict__ outt) {
  extern __shared__ __align__(16) char smem[];
  __bf16* tiles = (__bf16*)smem;  // [4][128*40]: L0, L1, R0, R1

  const int t = threadIdx.x, lane = t & 31, w = t >> 5;
  const int wi = w >> 2, wj = w & 3;
  const int c = blockIdx.z;
  const int i0 = blockIdx.y * 128, j0 = blockIdx.x * 128;
  const __bf16* Lb = left + (size_t)c * NPOS;
  const __bf16* Rb = right + (size_t)c * NPOS;

  const int srow = t >> 2;        // 0..63
  const int sq = t & 3;           // k-chunk of 8

  f32x8 acc[4][2];
#pragma unroll
  for (int si = 0; si < 4; ++si)
#pragma unroll
    for (int sj = 0; sj < 2; ++sj) acc[si][sj] = zero8();

  // Issue one K-slab (32 k) into buffer `buf`: 4 async DMA per thread.
  auto issue = [&](int ks, int buf) {
    const int k0 = ks * 32;
    __bf16* dL = tiles + buf * 5120;
    __bf16* dR = tiles + (2 + buf) * 5120;
#pragma unroll
    for (int rr = 0; rr < 2; ++rr) {
      int row = srow + rr * 64;
      async_ld16(dL + row * 40 + sq * 8,
                 Lb + (size_t)(i0 + row) * 768 + k0 + sq * 8);
      async_ld16(dR + row * 40 + sq * 8,
                 Rb + (size_t)(j0 + row) * 768 + k0 + sq * 8);
    }
  };

  issue(0, 0);
  wait_async0();
  __syncthreads();

  for (int ks = 0; ks < 24; ++ks) {
    const int cur = ks & 1;
    if (ks + 1 < 24) issue(ks + 1, 1 - cur);  // DMA next slab behind compute

    const __bf16* sL = tiles + cur * 5120;
    const __bf16* sR = tiles + (2 + cur) * 5120;

    bf16x16 bF[2];
#pragma unroll
    for (int sj = 0; sj < 2; ++sj)
      bF[sj] = load_fragB(sR + (wj * 32 + sj * 16 + (lane & 15)) * 40, lane);
#pragma unroll
    for (int si = 0; si < 4; ++si) {
      bf16x16 aF = load_fragA(sL + (wi * 64 + si * 16 + (lane & 15)) * 40, lane);
#pragma unroll
      for (int sj = 0; sj < 2; ++sj)
        acc[si][sj] = wmma_bf16(aF, bF[sj], acc[si][sj]);
    }

    if (ks + 1 < 24) {
      wait_async0();     // next slab landed (this wave's DMA)
      __syncthreads();   // all waves: slab ks+1 visible, compute(ks) retired
    }
  }

  float* Ob = outt + (size_t)c * NPOS;
#pragma unroll
  for (int si = 0; si < 4; ++si)
#pragma unroll
    for (int sj = 0; sj < 2; ++sj) {
      int jg = j0 + wj * 32 + sj * 16 + (lane & 15);
#pragma unroll
      for (int v = 0; v < 8; ++v) {
        int ig = i0 + wi * 64 + si * 16 + v + ((lane >> 4) * 8);
        Ob[(size_t)ig * 768 + jg] = acc[si][sj][v];
      }
    }
}

// ---------------------------------------------------------------------------
// Kernel 4: fused center-LayerNorm (over c) + output projection (WMMA) +
// gate multiply.  Block = 128 positions.  LN stats via LDS, normalized tile
// kept in LDS as bf16 [p][c] -> A operand; B = Wo rows [o][c].
// D[p][o] stores straight into d_out (o contiguous across lanes).
// ---------------------------------------------------------------------------
__global__ void k4_finalize(const float* __restrict__ outt,
                            const float* __restrict__ cnw,
                            const float* __restrict__ cnb,
                            const __bf16* __restrict__ wo,
                            const float* __restrict__ ob,
                            const __bf16* __restrict__ gate,
                            float* __restrict__ dout) {
  extern __shared__ __align__(16) char smem[];
  float* raw   = (float*)smem;                       // 128 x 132 f32
  __bf16* nrm  = (__bf16*)(smem + 128 * 132 * 4);    // 128 x 136 bf16
  __bf16* swo  = nrm + 128 * 136;                    // 128 x 136 bf16
  float* stats = (float*)(smem + 128 * 132 * 4 + 2 * 128 * 136 * 2); // 256 f32
  float* ps    = stats + 256;                        // 512 f32

  const int t = threadIdx.x, lane = t & 31, w = t >> 5;
  const size_t p0 = (size_t)blockIdx.x * 128;

  // stage Wo (128x128 bf16)
#pragma unroll
  for (int it = 0; it < 8; ++it) {
    int cidx = it * 256 + t;
    int row = cidx >> 3, q = cidx & 7;
    *(bf16x8*)(swo + row * 136 + q * 16) =
        *(const bf16x8*)(wo + row * 128 + q * 16);
  }

  // load out_t columns + partial stats; thread owns (p = t&127, half of c)
  const int pl = t & 127, chalf = t >> 7;
  float s1 = 0.0f, s2 = 0.0f;
#pragma unroll 4
  for (int cc = 0; cc < 64; ++cc) {
    int c = chalf * 64 + cc;
    float v = outt[(size_t)c * NPOS + p0 + pl];
    raw[pl * 132 + c] = v;
    s1 += v;
    s2 += v * v;
  }
  ps[t * 2] = s1;
  ps[t * 2 + 1] = s2;
  __syncthreads();
  if (t < 128) {
    float a1 = ps[t * 2] + ps[(t + 128) * 2];
    float a2 = ps[t * 2 + 1] + ps[(t + 128) * 2 + 1];
    float mu = a1 * (1.0f / 128.0f);
    float var = a2 * (1.0f / 128.0f) - mu * mu;
    stats[t * 2] = mu;
    stats[t * 2 + 1] = rsqrtf(var + 1e-5f);
  }
  __syncthreads();
  {
    float mu = stats[pl * 2], rstd = stats[pl * 2 + 1];
#pragma unroll 4
    for (int cc = 0; cc < 64; ++cc) {
      int c = chalf * 64 + cc;
      nrm[pl * 136 + c] =
          (__bf16)((raw[pl * 132 + c] - mu) * rstd * cnw[c] + cnb[c]);
    }
  }
  __syncthreads();

  // GEMM: wave w owns rows p0 + w*16 .. +15; N = 128 outputs (8 subtiles)
  bf16x16 aF[4];
#pragma unroll
  for (int ks = 0; ks < 4; ++ks)
    aF[ks] = load_fragA(nrm + (w * 16 + (lane & 15)) * 136 + ks * 32, lane);

  f32x8 acc[8];
#pragma unroll
  for (int ns = 0; ns < 8; ++ns) {
    acc[ns] = zero8();
#pragma unroll
    for (int ks = 0; ks < 4; ++ks) {
      bf16x16 bF = load_fragB(swo + (ns * 16 + (lane & 15)) * 136 + ks * 32, lane);
      acc[ns] = wmma_bf16(aF[ks], bF, acc[ns]);
    }
  }

  const int prow = w * 16 + ((lane >> 4) * 8);
#pragma unroll
  for (int ns = 0; ns < 8; ++ns) {
    int o = ns * 16 + (lane & 15);
    float bo = ob[o];
    const __bf16* gp = gate + (size_t)o * NPOS + p0 + prow;
#pragma unroll
    for (int v = 0; v < 8; ++v) {
      float g = (float)gp[v];
      dout[(p0 + prow + v) * 128 + o] = (acc[ns][v] + bo) * g;
    }
  }
}

// ---------------------------------------------------------------------------
extern "C" void kernel_launch(void* const* d_in, const int* in_sizes, int n_in,
                              void* d_out, int out_size, void* d_ws, size_t ws_size,
                              hipStream_t stream) {
  const float* act = (const float*)d_in[0];
  const float* mask = (const float*)d_in[1];
  const float* lnw = (const float*)d_in[2];
  const float* lnb = (const float*)d_in[3];
  const float* pw = (const float*)d_in[4];
  const float* pb = (const float*)d_in[5];
  const float* gw = (const float*)d_in[6];
  const float* gb = (const float*)d_in[7];
  const float* cnw = (const float*)d_in[8];
  const float* cnb = (const float*)d_in[9];
  const float* ow = (const float*)d_in[10];
  const float* obias = (const float*)d_in[11];
  const float* glw = (const float*)d_in[12];
  const float* glb = (const float*)d_in[13];
  float* out = (float*)d_out;

  const size_t S = (size_t)NPOS * 128 * 2;  // 150,994,944 bytes
  char* ws = (char*)d_ws;
  __bf16* x_bf   = (__bf16*)(ws);
  __bf16* left_t = (__bf16*)(ws + S);
  __bf16* right_t= (__bf16*)(ws + 2 * S);
  __bf16* gate_t = (__bf16*)(ws + 3 * S);
  float*  out_t  = (float*)(ws + 4 * S);
  __bf16* wcat   = (__bf16*)(ws + 6 * S);
  float*  bpk    = (float*)(ws + 6 * S + 163840);
  __bf16* wo_bf  = (__bf16*)(ws + 6 * S + 166400);

  k0_pack<<<(640 * 128 + 128 * 128 + 255) / 256, 256, 0, stream>>>(
      pw, pb, gw, gb, glw, glb, ow, wcat, bpk, wo_bf);

  k1_layernorm<<<NPOS / 8, 256, 0, stream>>>(act, lnw, lnb, x_bf);

  k2_project<<<NPOS / 128, 256, (160 * 136 + 128 * 136) * 2, stream>>>(
      x_bf, wcat, bpk, mask, left_t, right_t, gate_t);

  dim3 g3(NRES / 128, NRES / 128, CZ);
  k3_einsum<<<g3, 256, 4 * 128 * 40 * 2, stream>>>(left_t, right_t, out_t);

  size_t smem4 = 128 * 132 * 4 + 2 * 128 * 136 * 2 + 256 * 4 + 512 * 4;
  k4_finalize<<<NPOS / 128, 256, smem4, stream>>>(
      out_t, cnw, cnb, wo_bf, obias, gate_t, out);
}
